// CMSABlock_35081292873774
// MI455X (gfx1250) — matrix operand order repaired and checked by reading, not compiled, not run on hardware
//
#include <hip/hip_runtime.h>
#include <cstdint>
#include <cstddef>

// ---------------------------------------------------------------- constants
static constexpr int Bn      = 2;        // batch
static constexpr int Hh      = 64;
static constexpr int Ww      = 64;
static constexpr int Ll      = Hh * Ww;  // 4096
static constexpr int DMODEL  = 96;
static constexpr int DSTATE  = 16;
static constexpr int DTRANK  = 6;
static constexpr int Ee      = 192;
static constexpr int E2      = 384;      // 2*E
static constexpr int SD      = 4;        // scan directions
static constexpr int STn     = 2;        // streams
static constexpr int PARAMS  = DTRANK + 2 * DSTATE;  // 38
static constexpr int PPAD    = 48;       // PARAMS padded to N-tile multiple
static constexpr int MSL     = STn * Ll; // 8192

// ---------------------------------------------------------------- types
typedef unsigned short bf16_t;
typedef __attribute__((ext_vector_type(16))) __bf16        bf16x16;
typedef __attribute__((ext_vector_type(8)))  float         f32x8;
typedef __attribute__((ext_vector_type(4)))  unsigned int  u32x4;

__device__ __forceinline__ bf16_t f2bf(float f) {
    unsigned u = __float_as_uint(f);
    u += 0x7FFFu + ((u >> 16) & 1u);            // round-to-nearest-even
    return (bf16_t)(u >> 16);
}

union FragU { u32x4 u[2]; bf16x16 v; };

// A fragment (16x32 bf16): lane<16 holds row M=lane, K = kb+{0..7, 16..23};
// lane>=16 holds row M=lane-16, K = kb+{8..15, 24..31}. rowPtr = &A[row*lda].
__device__ __forceinline__ bf16x16 ld_afrag(const bf16_t* rowPtr, int kb, int half) {
    FragU t;
    const bf16_t* p = rowPtr + kb + half * 8;
    t.u[0] = *(const u32x4*)(p);
    t.u[1] = *(const u32x4*)(p + 16);
    return t.v;
}
// B fragment (32x16 bf16, B stored transposed: N-major, K contiguous):
// lane<16 holds col N=lane, K = kb+{0..7, 16..23}; lane>=16 holds K = kb+{8..15, 24..31}.
__device__ __forceinline__ bf16x16 ld_bfrag(const bf16_t* colPtr, int kb, int half) {
    FragU t;
    const bf16_t* p = colPtr + kb + half * 8;
    t.u[0] = *(const u32x4*)(p);
    t.u[1] = *(const u32x4*)(p + 16);
    return t.v;
}

// cross-scan index map: sequence position l (direction d) -> spatial index
__device__ __forceinline__ int map_sp(int d, int l) {
    int ll = (d >= 2) ? (Ll - 1 - l) : l;
    return (d & 1) ? (((ll & 63) << 6) | (ll >> 6)) : ll;
}

// ---------------------------------------------------------------- small utils
__global__ void cvt_bf16_kernel(const float* __restrict__ src,
                                bf16_t* __restrict__ dst, int n) {
    int i = blockIdx.x * blockDim.x + threadIdx.x;
    if (i < n) dst[i] = f2bf(src[i]);
}

// pack x_proj_weight (SD,ST,38,192) f32 -> (SD,ST,48,192) bf16, zero-padded
__global__ void pack_xproj_kernel(const float* __restrict__ src,
                                  bf16_t* __restrict__ dst) {
    int i = blockIdx.x * blockDim.x + threadIdx.x;
    if (i >= SD * STn * PPAD * Ee) return;
    int c = i % Ee;
    int p = (i / Ee) % PPAD;
    int s = (i / (Ee * PPAD)) % STn;
    int d = i / (Ee * PPAD * STn);
    dst[i] = (p < PARAMS) ? f2bf(src[((d * STn + s) * PARAMS + p) * Ee + c]) : (bf16_t)0;
}

// LayerNorm over 96, write bf16 rows
__global__ __launch_bounds__(96) void ln96_kernel(const float* __restrict__ x,
                                                  const float* __restrict__ w,
                                                  const float* __restrict__ b,
                                                  bf16_t* __restrict__ out) {
    __shared__ float red[96];
    const int row = blockIdx.x, t = threadIdx.x;
    float v = x[row * 96 + t];
    red[t] = v; __syncthreads();
    for (int s = 48; s >= 3; s >>= 1) { if (t < s) red[t] += red[t + s]; __syncthreads(); }
    float mu = (red[0] + red[1] + red[2]) * (1.0f / 96.0f);
    __syncthreads();
    float dvi = v - mu;
    red[t] = dvi * dvi; __syncthreads();
    for (int s = 48; s >= 3; s >>= 1) { if (t < s) red[t] += red[t + s]; __syncthreads(); }
    float var = (red[0] + red[1] + red[2]) * (1.0f / 96.0f);
    out[row * 96 + t] = f2bf(dvi * rsqrtf(var + 1e-5f) * w[t] + b[t]);
}

// ---------------------------------------------------------------- GEMM 1
// C(8192,384) = LN(x) (8192x96 bf16) @ w_in^T ; store f32 NCHW (B,384,L)
__global__ __launch_bounds__(32) void gemm1_kernel(const bf16_t* __restrict__ A,
                                                   const bf16_t* __restrict__ Bt,
                                                   float* __restrict__ hbuf) {
    const int lane = threadIdx.x & 31, half = lane >> 4;
    const int m0 = blockIdx.x * 16, n0 = blockIdx.y * 16;
    const bf16_t* aRow = A + (m0 + (lane & 15)) * DMODEL;
    const bf16_t* bCol = Bt + (n0 + (lane & 15)) * DMODEL;
    f32x8 acc = {};
#pragma unroll
    for (int kb = 0; kb < DMODEL; kb += 32) {
        bf16x16 a = ld_afrag(aRow, kb, half);
        bf16x16 b = ld_bfrag(bCol, kb, half);
        acc = __builtin_amdgcn_wmma_f32_16x16x32_bf16(false, a, false, b,
                                                      (short)0, acc, false, false);
    }
    const int col = n0 + (lane & 15);
#pragma unroll
    for (int r = 0; r < 8; ++r) {
        int m = m0 + r + half * 8;
        int bb = m >> 12, l = m & (Ll - 1);
        hbuf[(bb * E2 + col) * Ll + l] = acc[r];
    }
}

// ---------------------------------------------------------------- conv
// depthwise 3x3 SAME + bias + SiLU; f32 NCHW in -> bf16 (B*L,384) out
__global__ void dwconv_silu_kernel(const float* __restrict__ hin,
                                   const float* __restrict__ cw,
                                   const float* __restrict__ cb,
                                   bf16_t* __restrict__ outbf) {
    int idx = blockIdx.x * blockDim.x + threadIdx.x;
    if (idx >= Bn * E2 * Ll) return;
    int hw = idx & (Ll - 1);
    int c  = (idx >> 12) % E2;
    int b  = idx / (E2 * Ll);
    int hh = hw >> 6, ww = hw & 63;
    float acc = cb[c];
    const float* base = hin + (b * E2 + c) * Ll;
#pragma unroll
    for (int ky = 0; ky < 3; ++ky) {
        int y = hh + ky - 1;
        if ((unsigned)y >= (unsigned)Hh) continue;
#pragma unroll
        for (int kx = 0; kx < 3; ++kx) {
            int x = ww + kx - 1;
            if ((unsigned)x >= (unsigned)Ww) continue;
            acc = fmaf(base[y * Ww + x], cw[c * 9 + ky * 3 + kx], acc);
        }
    }
    float sv = acc / (1.0f + __expf(-acc));   // SiLU
    outbf[(b * Ll + hw) * E2 + c] = f2bf(sv);
}

// ---------------------------------------------------------------- GEMM 2
// C(8192,192) = conv (8192x384 bf16) @ pmg^T + pmg_b; write f32 + bf16 (B,ST,L,E)
__global__ __launch_bounds__(32) void gemm2_kernel(const bf16_t* __restrict__ A,
                                                   const bf16_t* __restrict__ Bt,
                                                   const float* __restrict__ bias,
                                                   float* __restrict__ xf32,
                                                   bf16_t* __restrict__ xbf,
                                                   int s) {
    const int lane = threadIdx.x & 31, half = lane >> 4;
    const int m0 = blockIdx.x * 16, n0 = blockIdx.y * 16;
    const bf16_t* aRow = A + (m0 + (lane & 15)) * E2;
    const bf16_t* bCol = Bt + (n0 + (lane & 15)) * E2;
    f32x8 acc = {};
#pragma unroll
    for (int kb = 0; kb < E2; kb += 32) {
        bf16x16 a = ld_afrag(aRow, kb, half);
        bf16x16 b = ld_bfrag(bCol, kb, half);
        acc = __builtin_amdgcn_wmma_f32_16x16x32_bf16(false, a, false, b,
                                                      (short)0, acc, false, false);
    }
    const int col = n0 + (lane & 15);
    const float bv = bias[col];
#pragma unroll
    for (int r = 0; r < 8; ++r) {
        int m = m0 + r + half * 8;
        int bb = m >> 12, l = m & (Ll - 1);
        size_t o = ((size_t)(bb * STn + s) * Ll + l) * Ee + col;
        float v = acc[r] + bv;
        xf32[o] = v;
        xbf[o]  = f2bf(v);
    }
}

// ---------------------------------------------------------------- GEMM 3
// per (b,d,s): x_dbl(l,p) = xs_row(l) (gathered, 192 bf16) @ xproj^T (48x192)
__global__ __launch_bounds__(32) void gemm3_kernel(const bf16_t* __restrict__ xbf,
                                                   const bf16_t* __restrict__ wproj,
                                                   float* __restrict__ xdbl) {
    const int lane = threadIdx.x & 31, half = lane >> 4;
    const int z = blockIdx.z;
    const int b = z >> 3, d = (z >> 1) & 3, s = z & 1;
    const int m0 = blockIdx.x * 16, n0 = blockIdx.y * 16;
    const int l_lane = m0 + (lane & 15);
    const int sp = map_sp(d, l_lane);
    const bf16_t* aRow = xbf + ((size_t)(b * STn + s) * Ll + sp) * Ee;
    const bf16_t* bCol = wproj + ((size_t)(d * STn + s) * PPAD + n0 + (lane & 15)) * Ee;
    f32x8 acc = {};
#pragma unroll
    for (int kb = 0; kb < Ee; kb += 32) {
        bf16x16 a = ld_afrag(aRow, kb, half);
        bf16x16 b_ = ld_bfrag(bCol, kb, half);
        acc = __builtin_amdgcn_wmma_f32_16x16x32_bf16(false, a, false, b_,
                                                      (short)0, acc, false, false);
    }
    const int col = n0 + (lane & 15);
    const size_t rowbase = ((size_t)((b * SD + d) * STn + s) * Ll);
#pragma unroll
    for (int r = 0; r < 8; ++r) {
        int l = m0 + r + half * 8;
        xdbl[(rowbase + l) * PPAD + col] = acc[r];
    }
}

// ---------------------------------------------------------------- selective scan
// one block per (b,d); one lane per channel c; 16 states in VGPRs
__global__ __launch_bounds__(192) void scan_kernel(const float* __restrict__ xdbl,
                                                   const float* __restrict__ u_,
                                                   const float* __restrict__ dtw,
                                                   const float* __restrict__ dtb,
                                                   const float* __restrict__ A_logs,
                                                   const float* __restrict__ Ds,
                                                   float* __restrict__ ybuf) {
    const int blk = blockIdx.x;
    const int b = blk >> 2, d = blk & 3;
    const int c = threadIdx.x;
    const int ch = d * Ee + c;

    float An[DSTATE];
#pragma unroll
    for (int n = 0; n < DSTATE; ++n) An[n] = -__expf(A_logs[ch * DSTATE + n]);
    const float Dc = Ds[ch];
    const float bias = dtb[ch];
    float w0[DTRANK], w1[DTRANK];
#pragma unroll
    for (int r = 0; r < DTRANK; ++r) {
        w0[r] = dtw[((0 * SD + d) * Ee + c) * DTRANK + r];
        w1[r] = dtw[((1 * SD + d) * Ee + c) * DTRANK + r];
    }
    float h[DSTATE];
#pragma unroll
    for (int n = 0; n < DSTATE; ++n) h[n] = 0.0f;

    const size_t dblbase = (size_t)(b * SD + d) * STn * Ll;      // rows of xdbl
    for (int m = 0; m < MSL; ++m) {
        const int s = m & 1, l = m >> 1;
        const int sp = map_sp(d, l);
        const float* xr = xdbl + (dblbase + (size_t)s * Ll + l) * PPAD;
        __builtin_prefetch(xr + PPAD, 0, 1);   // global_prefetch next param row
        float dt = bias;
        if (s) {
#pragma unroll
            for (int r = 0; r < DTRANK; ++r) dt = fmaf(xr[r], w1[r], dt);
        } else {
#pragma unroll
            for (int r = 0; r < DTRANK; ++r) dt = fmaf(xr[r], w0[r], dt);
        }
        float delta = (dt > 20.0f) ? dt : log1pf(__expf(dt));    // softplus
        float uu = u_[((size_t)(b * STn + s) * Ll + sp) * Ee + c];
        float du = delta * uu;
        float acc = 0.0f;
#pragma unroll
        for (int n = 0; n < DSTATE; ++n) {
            float dA = __expf(delta * An[n]);
            h[n] = fmaf(dA, h[n], du * xr[DTRANK + n]);
            acc  = fmaf(h[n], xr[DTRANK + DSTATE + n], acc);
        }
        // scatter to spatial order: (b,s,sp,d,c) — map is an involution
        ybuf[(((size_t)(b * STn + s) * Ll + sp) * SD + d) * Ee + c] = acc + uu * Dc;
    }
}

// ---------------------------------------------------------------- merge + LN(192)
__global__ __launch_bounds__(192) void merge_ln_kernel(const float* __restrict__ ybuf,
                                                       const float* __restrict__ w,
                                                       const float* __restrict__ bb,
                                                       bf16_t* __restrict__ out) {
    __shared__ float red[Ee];
    const int row = blockIdx.x, t = threadIdx.x;
    const float* p = ybuf + (size_t)row * SD * Ee;
    float v = p[t] + p[Ee + t] + p[2 * Ee + t] + p[3 * Ee + t];
    red[t] = v; __syncthreads();
    for (int s = 96; s >= 3; s >>= 1) { if (t < s) red[t] += red[t + s]; __syncthreads(); }
    float mu = (red[0] + red[1] + red[2]) * (1.0f / (float)Ee);
    __syncthreads();
    float dv = v - mu;
    red[t] = dv * dv; __syncthreads();
    for (int s = 96; s >= 3; s >>= 1) { if (t < s) red[t] += red[t + s]; __syncthreads(); }
    float var = (red[0] + red[1] + red[2]) * (1.0f / (float)Ee);
    out[(size_t)row * Ee + t] = f2bf(dv * rsqrtf(var + 1e-5f) * w[t] + bb[t]);
}

// ---------------------------------------------------------------- GEMM 5 (output)
// C(16384,96) = merged (16384x192 bf16) @ w_out^T -> d_out f32
__global__ __launch_bounds__(32) void gemm5_kernel(const bf16_t* __restrict__ A,
                                                   const bf16_t* __restrict__ Bt,
                                                   float* __restrict__ out) {
    const int lane = threadIdx.x & 31, half = lane >> 4;
    const int m0 = blockIdx.x * 16, n0 = blockIdx.y * 16;
    const bf16_t* aRow = A + (size_t)(m0 + (lane & 15)) * Ee;
    const bf16_t* bCol = Bt + (size_t)(n0 + (lane & 15)) * Ee;
    f32x8 acc = {};
#pragma unroll
    for (int kb = 0; kb < Ee; kb += 32) {
        bf16x16 a = ld_afrag(aRow, kb, half);
        bf16x16 b = ld_bfrag(bCol, kb, half);
        acc = __builtin_amdgcn_wmma_f32_16x16x32_bf16(false, a, false, b,
                                                      (short)0, acc, false, false);
    }
    const int col = n0 + (lane & 15);
#pragma unroll
    for (int r = 0; r < 8; ++r) {
        int m = m0 + r + half * 8;
        out[(size_t)m * DMODEL + col] = acc[r];
    }
}

// ---------------------------------------------------------------- launcher
extern "C" void kernel_launch(void* const* d_in, const int* in_sizes, int n_in,
                              void* d_out, int out_size, void* d_ws, size_t ws_size,
                              hipStream_t stream) {
    (void)in_sizes; (void)n_in; (void)out_size; (void)ws_size;

    const float* x0    = (const float*)d_in[0];
    const float* x1    = (const float*)d_in[1];
    const float* ln0w  = (const float*)d_in[2];
    const float* ln0b  = (const float*)d_in[3];
    const float* ln1w  = (const float*)d_in[4];
    const float* ln1b  = (const float*)d_in[5];
    const float* win0  = (const float*)d_in[6];
    const float* win1  = (const float*)d_in[7];
    const float* convw = (const float*)d_in[8];
    const float* convb = (const float*)d_in[9];
    const float* pmgw  = (const float*)d_in[10];
    const float* pmgb  = (const float*)d_in[11];
    const float* xprjw = (const float*)d_in[12];
    const float* dtw   = (const float*)d_in[13];
    const float* dtb   = (const float*)d_in[14];
    const float* Alogs = (const float*)d_in[15];
    const float* Dsp   = (const float*)d_in[16];
    const float* onw   = (const float*)d_in[17];
    const float* onb   = (const float*)d_in[18];
    const float* wout  = (const float*)d_in[19];
    float* outp = (float*)d_out;

    // bump allocator over ws (all chunks 256B aligned)
    char* base = (char*)d_ws;
    size_t off = 0;
    auto alloc = [&](size_t bytes) -> char* {
        char* p = base + off;
        off += (bytes + 255) & ~(size_t)255;
        return p;
    };
    const int ROWS1 = Bn * Ll;          // 8192
    bf16_t* lnbuf0  = (bf16_t*)alloc((size_t)ROWS1 * DMODEL * 2);
    bf16_t* lnbuf1  = (bf16_t*)alloc((size_t)ROWS1 * DMODEL * 2);
    bf16_t* winbf0  = (bf16_t*)alloc((size_t)E2 * DMODEL * 2);
    bf16_t* winbf1  = (bf16_t*)alloc((size_t)E2 * DMODEL * 2);
    bf16_t* pmgbf   = (bf16_t*)alloc((size_t)Ee * E2 * 2);
    bf16_t* woutbf  = (bf16_t*)alloc((size_t)DMODEL * Ee * 2);
    bf16_t* xprojbf = (bf16_t*)alloc((size_t)SD * STn * PPAD * Ee * 2);
    float*  hbuf0   = (float*)alloc((size_t)Bn * E2 * Ll * 4);
    float*  hbuf1   = (float*)alloc((size_t)Bn * E2 * Ll * 4);
    bf16_t* convbf0 = (bf16_t*)alloc((size_t)ROWS1 * E2 * 2);
    bf16_t* convbf1 = (bf16_t*)alloc((size_t)ROWS1 * E2 * 2);
    float*  xf32    = (float*)alloc((size_t)Bn * STn * Ll * Ee * 4);
    bf16_t* xbf     = (bf16_t*)alloc((size_t)Bn * STn * Ll * Ee * 2);
    float*  xdbl    = (float*)alloc((size_t)Bn * SD * STn * Ll * PPAD * 4);
    float*  ybuf    = (float*)alloc((size_t)Bn * STn * Ll * SD * Ee * 4);
    bf16_t* mergebf = (bf16_t*)alloc((size_t)Bn * STn * Ll * Ee * 2);

    // weight conversions
    auto cvt = [&](const float* s, bf16_t* dd, int n) {
        cvt_bf16_kernel<<<(n + 255) / 256, 256, 0, stream>>>(s, dd, n);
    };
    cvt(win0, winbf0, E2 * DMODEL);
    cvt(win1, winbf1, E2 * DMODEL);
    cvt(pmgw, pmgbf, Ee * E2);
    cvt(wout, woutbf, DMODEL * Ee);
    pack_xproj_kernel<<<(SD * STn * PPAD * Ee + 255) / 256, 256, 0, stream>>>(xprjw, xprojbf);

    // per-stream front end
    const float* xs[2]   = { x0, x1 };
    const float* lws[2]  = { ln0w, ln1w };
    const float* lbs[2]  = { ln0b, ln1b };
    bf16_t* lnb[2]       = { lnbuf0, lnbuf1 };
    bf16_t* wib[2]       = { winbf0, winbf1 };
    float*  hb[2]        = { hbuf0, hbuf1 };
    bf16_t* cvb[2]       = { convbf0, convbf1 };
    for (int s = 0; s < STn; ++s) {
        ln96_kernel<<<ROWS1, 96, 0, stream>>>(xs[s], lws[s], lbs[s], lnb[s]);
        gemm1_kernel<<<dim3(ROWS1 / 16, E2 / 16), 32, 0, stream>>>(lnb[s], wib[s], hb[s]);
        dwconv_silu_kernel<<<(Bn * E2 * Ll + 255) / 256, 256, 0, stream>>>(hb[s], convw, convb, cvb[s]);
        gemm2_kernel<<<dim3(ROWS1 / 16, Ee / 16), 32, 0, stream>>>(cvb[s], pmgbf, pmgb, xf32, xbf, s);
    }

    // cross-scan projection (16 (b,d,s) combos, M=4096, N=48, K=192)
    gemm3_kernel<<<dim3(Ll / 16, PPAD / 16, Bn * SD * STn), 32, 0, stream>>>(xbf, xprojbf, xdbl);

    // selective scan: 8 blocks of 192 lanes, 16 states each in VGPRs
    scan_kernel<<<Bn * SD, Ee, 0, stream>>>(xdbl, xf32, dtw, dtb, Alogs, Dsp, ybuf);

    // cross-merge + out-LayerNorm
    merge_ln_kernel<<<Bn * STn * Ll, Ee, 0, stream>>>(ybuf, onw, onb, mergebf);

    // output projection
    gemm5_kernel<<<dim3((Bn * STn * Ll) / 16, DMODEL / 16), 32, 0, stream>>>(mergebf, woutbf, outp);
}
